// MultiHeadAttentionSpatial_68496138436653
// MI455X (gfx1250) — matrix operand order repaired and checked by reading, not compiled, not run on hardware
//
#include <hip/hip_runtime.h>
#include <hip/hip_bf16.h>

// ---------------------------------------------------------------------------
// MultiHeadAttentionSpatial for MI455X (gfx1250), bf16 WMMA everywhere.
// ---------------------------------------------------------------------------

#define D_MODEL 768
#define N_HEAD  12
#define DH      64
#define SD      5
#define BATCH   16
#define SEQ     512
#define MROWS   (BATCH * SEQ)          // 8192

typedef __attribute__((ext_vector_type(16))) __bf16 bf16x16;
typedef __attribute__((ext_vector_type(8)))  float  f32x8;

struct alignas(16) U4 { unsigned x, y, z, w; };
struct alignas(8)  U2 { unsigned x, y; };
struct alignas(16) F4 { float x, y, z, w; };

union Frag {
    bf16x16 v;
    U4      q[2];
};

__device__ inline unsigned short f2bf(float f) {
    union { float f; unsigned u; } c; c.f = f;
    unsigned r = 0x7FFFu + ((c.u >> 16) & 1u);      // round-to-nearest-even
    return (unsigned short)((c.u + r) >> 16);
}
__device__ inline unsigned pack2bf(float lo, float hi) {
    return (unsigned)f2bf(lo) | ((unsigned)f2bf(hi) << 16);
}

// --- CDNA5 async global->LDS copies (ASYNCcnt path), guarded --------------
#if defined(__HIP_DEVICE_COMPILE__) && \
    __has_builtin(__builtin_amdgcn_global_load_async_to_lds_b128) && \
    __has_builtin(__builtin_amdgcn_s_wait_asynccnt)
#define HAVE_ASYNC_LDS 1
typedef int v4i_t __attribute__((vector_size(16)));            // 16-byte granule
typedef __attribute__((address_space(1))) v4i_t* gas_ptr;      // global (AS1)
typedef __attribute__((address_space(3))) v4i_t* las_ptr;      // LDS (AS3)
__device__ inline gas_ptr to_gas(const void* p) {
    return (gas_ptr)(unsigned long long)p;
}
__device__ inline las_ptr to_las(void* p) {
    return (las_ptr)(unsigned)(unsigned long long)p;   // low 32 bits = LDS addr
}
#else
#define HAVE_ASYNC_LDS 0
#endif

// ===========================================================================
// GEMM: out[8192,768] = A[8192,768] @ W[768,768]^T + bias (+ residual)
// 64x64 tile per 256-thread block (8 waves, 2 WMMA tiles each), K-step 32.
// LDS rows padded to 40 ushorts (80 B = odd multiple of 16 B): aligned b128
// fragment loads, 16 lanes -> 16 distinct banks. Templated epilogue =>
// straight-line stores. global_prefetch hides the next K-tile's HBM latency.
// ===========================================================================
#define LDP 40

template<bool BF16OUT, bool RES>
__global__ __launch_bounds__(256) void proj_gemm_kernel(
    const float* __restrict__ A, const float* __restrict__ W,
    const float* __restrict__ bias, const float* __restrict__ residual,
    float* __restrict__ outF, unsigned short* __restrict__ outB)
{
    __shared__ unsigned short Alds[64 * LDP];
    __shared__ unsigned short Blds[64 * LDP];

    const int tid  = threadIdx.x;
    const int lane = tid & 31;
    const int wid  = tid >> 5;
    const int half = lane >> 4;
    const int l15  = lane & 15;
    const int m0   = blockIdx.y * 64;
    const int n0   = blockIdx.x * 64;

    f32x8 acc0 = {};
    f32x8 acc1 = {};

    for (int kk = 0; kk < D_MODEL; kk += 32) {
        // 64x32 tiles of A and W -> bf16 in LDS (float4 loads, 8B LDS stores)
        for (int idx = tid; idx < 64 * 8; idx += 256) {
            int r = idx >> 3, c4 = (idx & 7) * 4;
            const float* pa = &A[(size_t)(m0 + r) * D_MODEL + kk + c4];
            const float* pw = &W[(size_t)(n0 + r) * D_MODEL + kk + c4];
            F4 fa = *(const F4*)pa;
            F4 fw = *(const F4*)pw;
            if (kk + 32 < D_MODEL) {                    // prefetch next K tile
                __builtin_prefetch(pa + 32, 0, 1);
                __builtin_prefetch(pw + 32, 0, 1);
            }
            U2 ua; ua.x = pack2bf(fa.x, fa.y); ua.y = pack2bf(fa.z, fa.w);
            U2 uw; uw.x = pack2bf(fw.x, fw.y); uw.y = pack2bf(fw.z, fw.w);
            *(U2*)&Alds[r * LDP + c4] = ua;
            *(U2*)&Blds[r * LDP + c4] = uw;
        }
        __syncthreads();

        #pragma unroll
        for (int i = 0; i < 2; ++i) {
            int ti = wid * 2 + i;
            int mt = ti >> 2, nt = ti & 3;
            Frag a, b;
            const unsigned short* ap = &Alds[(mt * 16 + l15) * LDP + half * 8];
            a.q[0] = *(const U4*)ap;
            a.q[1] = *(const U4*)(ap + 16);
            const unsigned short* bp = &Blds[(nt * 16 + l15) * LDP + half * 8];
            b.q[0] = *(const U4*)bp;
            b.q[1] = *(const U4*)(bp + 16);
            if (i == 0)
                acc0 = __builtin_amdgcn_wmma_f32_16x16x32_bf16(false, a.v, false, b.v,
                                                               (short)0, acc0, false, false);
            else
                acc1 = __builtin_amdgcn_wmma_f32_16x16x32_bf16(false, a.v, false, b.v,
                                                               (short)0, acc1, false, false);
        }
        __syncthreads();
    }

    #pragma unroll
    for (int i = 0; i < 2; ++i) {
        int ti = wid * 2 + i;
        int mt = ti >> 2, nt = ti & 3;
        const int n = n0 + nt * 16 + l15;
        const float bv = bias[n];
        size_t idx = (size_t)(m0 + mt * 16 + half * 8) * D_MODEL + n;
        #pragma unroll
        for (int r = 0; r < 8; ++r, idx += D_MODEL) {
            float vv = (i == 0 ? acc0[r] : acc1[r]) + bv;
            if constexpr (RES)     vv += residual[idx];
            if constexpr (BF16OUT) outB[idx] = f2bf(vv);
            else                   outF[idx] = vv;
        }
    }
}

// ===========================================================================
// sw = (q + txt) @ Wl^T + bl    -> (b, l, 72)
// ===========================================================================
__global__ __launch_bounds__(96) void sw_kernel(
    const float* __restrict__ q, const float* __restrict__ txt,
    const float* __restrict__ Wl, const float* __restrict__ bl,
    float* __restrict__ sw)
{
    __shared__ float x[D_MODEL];
    const int row = blockIdx.x;            // b*512 + l
    const int b   = row >> 9;
    const int tid = threadIdx.x;
    for (int k = tid; k < D_MODEL; k += 96)
        x[k] = q[(size_t)row * D_MODEL + k] + txt[(size_t)b * D_MODEL + k];
    __syncthreads();
    if (tid < N_HEAD * (SD + 1)) {
        const float* w = &Wl[(size_t)tid * D_MODEL];
        float s = bl[tid];
        for (int k = 0; k < D_MODEL; ++k) s += x[k] * w[k];
        sw[(size_t)row * (N_HEAD * (SD + 1)) + tid] = s;
    }
}

// ===========================================================================
// Fused spatial attention. One workgroup per (batch, 16-query tile), looping
// over all 12 heads so pairwise_locs (160 KB slab in LDS) streams from HBM
// exactly once — staged with GLOBAL_LOAD_ASYNC_TO_LDS_B128 when available.
// All matmul fragments are 16B-aligned vector loads per the ISA layout
// tables. PV uses split-K across all 8 waves.
// ===========================================================================
#define SSTRIDE 520   // floats  (2080 B)
#define PSTRIDE 520   // ushorts (1040 B, odd multiple of 16 B)
#define VSTRIDE 520   // ushorts

#define LOCS_OFF 0
#define S_OFF    163840
#define P_OFF    (S_OFF + 16 * SSTRIDE * 4)          // 197120
#define V_OFF    (P_OFF + 16 * PSTRIDE * 2)          // 213760
#define RED_OFF  (V_OFF + 64 * VSTRIDE * 2)          // 280320
#define SB_OFF   (RED_OFF + 16 * 16 * 4)             // 281344
#define SWT_OFF  (SB_OFF + 64)                       // 281408
#define INV_OFF  (SWT_OFF + 16 * 8 * 4)              // 281920
#define MSK_OFF  (INV_OFF + 64)                      // 281984
#define SMEM_TOTAL (MSK_OFF + 512)                   // 282496 B  (< 320 KB)

__global__ __launch_bounds__(256) void attn_kernel(
    const unsigned short* __restrict__ Qb, const unsigned short* __restrict__ Kb,
    const unsigned short* __restrict__ Vb, const float* __restrict__ swbuf,
    const float* __restrict__ locs, const unsigned char* __restrict__ mask,
    float* __restrict__ O)
{
    extern __shared__ char smem[];
    float*          locs_s = (float*)(smem + LOCS_OFF);
    float*          S      = (float*)(smem + S_OFF);
    unsigned short* P      = (unsigned short*)(smem + P_OFF);
    unsigned short* Vt     = (unsigned short*)(smem + V_OFF);
    float*          red    = (float*)(smem + RED_OFF);
    float*          sbias  = (float*)(smem + SB_OFF);
    float*          swt    = (float*)(smem + SWT_OFF);
    float*          invs   = (float*)(smem + INV_OFF);
    unsigned char*  mrow   = (unsigned char*)(smem + MSK_OFF);

    const int tid  = threadIdx.x;
    const int lane = tid & 31;
    const int wid  = tid >> 5;
    const int half = lane >> 4;
    const int l15  = lane & 15;
    const int b    = blockIdx.y;
    const int l0   = blockIdx.x * 16;

    // --- stage the contiguous 160 KB loc slab + mask row for this tile ----
#if HAVE_ASYNC_LDS
    {
        gas_ptr src = to_gas(locs + ((size_t)(b * SEQ + l0) * SEQ) * SD);
        las_ptr dst = to_las(locs_s);
        for (int i = tid; i < (16 * SEQ * SD * 4) / 16; i += 256)
            __builtin_amdgcn_global_load_async_to_lds_b128(src + i, dst + i, 0, 0);
        if (wid == 0) {                               // wave-uniform, 512 B
            gas_ptr ms = to_gas(mask + (size_t)b * SEQ);
            las_ptr md = to_las(mrow);
            __builtin_amdgcn_global_load_async_to_lds_b128(ms + lane, md + lane, 0, 0);
        }
        __builtin_amdgcn_s_wait_asynccnt(0);
    }
#else
    {
        const F4* src = (const F4*)(locs + ((size_t)(b * SEQ + l0) * SEQ) * SD);
        F4* dst = (F4*)locs_s;
        for (int i = tid; i < (16 * SEQ * SD) / 4; i += 256) dst[i] = src[i];
        for (int i = tid; i < SEQ; i += 256) mrow[i] = mask[(size_t)b * SEQ + i];
    }
#endif

    for (int h = 0; h < N_HEAD; ++h) {
        // spatial gate params for this head / these 16 queries
        if (tid < 96) {
            int r = tid / 6, c = tid % 6;
            float vv = swbuf[(size_t)(b * SEQ + l0 + r) * (N_HEAD * 6) + h * 6 + c];
            if (c == 0) sbias[r] = vv; else swt[r * 8 + (c - 1)] = vv;
        }
        // V^T tile into LDS: Vt[feature][key]  (dword global loads, 2 feats)
        for (int i = tid; i < DH * SEQ / 2; i += 256) {
            int t = i >> 5, f2 = (i & 31) * 2;
            unsigned uu = *(const unsigned*)&Vb[(size_t)(b * SEQ + t) * D_MODEL + h * DH + f2];
            Vt[(f2 + 0) * VSTRIDE + t] = (unsigned short)(uu & 0xFFFFu);
            Vt[(f2 + 1) * VSTRIDE + t] = (unsigned short)(uu >> 16);
        }
        __syncthreads();

        // ---- S = Q K^T * 1/8, fused sigmoid-gate / log / mask ------------
        Frag qa[2];
        {
            const unsigned short* qrow = Qb + (size_t)(b * SEQ + l0 + l15) * D_MODEL + h * DH;
            #pragma unroll
            for (int kc = 0; kc < 2; ++kc) {
                const unsigned short* p = qrow + kc * 32 + half * 8;
                qa[kc].q[0] = *(const U4*)p;
                qa[kc].q[1] = *(const U4*)(p + 16);
            }
        }
        #pragma unroll
        for (int nt4 = 0; nt4 < 4; ++nt4) {
            int nt = wid * 4 + nt4;
            int tb = nt * 16;
            const unsigned short* krow = Kb + (size_t)(b * SEQ + tb + l15) * D_MODEL + h * DH;
            f32x8 acc = {};
            #pragma unroll
            for (int kc = 0; kc < 2; ++kc) {
                Frag kb;
                const unsigned short* p = krow + kc * 32 + half * 8;
                kb.q[0] = *(const U4*)p;
                kb.q[1] = *(const U4*)(p + 16);
                acc = __builtin_amdgcn_wmma_f32_16x16x32_bf16(false, qa[kc].v, false, kb.v,
                                                              (short)0, acc, false, false);
            }
            int t = tb + l15;
            bool msk = mrow[t] != 0;
            #pragma unroll
            for (int r = 0; r < 8; ++r) {
                int Mi = r + half * 8;
                float logit;
                if (msk) {
                    logit = -1.0e9f - 13.815511f;          // NEG_INF + log(1e-6)
                } else {
                    const float* lp = locs_s + ((size_t)Mi * SEQ + t) * SD;
                    const float* w  = swt + Mi * 8;
                    float dotv = sbias[Mi] + lp[0]*w[0] + lp[1]*w[1] + lp[2]*w[2]
                               + lp[3]*w[3] + lp[4]*w[4];
                    float sig = 1.f / (1.f + __expf(-dotv));
                    logit = __logf(fmaxf(sig, 1e-6f)) + acc[r] * 0.125f;
                }
                S[Mi * SSTRIDE + t] = logit;
            }
        }
        __syncthreads();

        // ---- row softmax (16 threads per row), P = exp(s-max) in bf16 ----
        {
            int row = tid >> 4, seg = tid & 15;
            float* srow = S + row * SSTRIDE + seg * 32;
            float mx = srow[0];
            for (int c = 1; c < 32; ++c) mx = fmaxf(mx, srow[c]);
            red[row * 16 + seg] = mx;
            __syncthreads();
            float m = red[row * 16];
            #pragma unroll
            for (int i = 1; i < 16; ++i) m = fmaxf(m, red[row * 16 + i]);
            __syncthreads();
            float sum = 0.f;
            unsigned short* prow = P + row * PSTRIDE + seg * 32;
            for (int c = 0; c < 32; ++c) {
                float e = __expf(srow[c] - m);
                sum += e;
                prow[c] = f2bf(e);
            }
            red[row * 16 + seg] = sum;
            __syncthreads();
            if (seg == 0) {
                float s = 0.f;
                #pragma unroll
                for (int i = 0; i < 16; ++i) s += red[row * 16 + i];
                invs[row] = 1.f / s;
            }
            __syncthreads();
        }

        // ---- O = P V : split-K over all 8 waves (S region reused) --------
        {
            const int ft = wid & 3;                 // 16-feature tile
            const int kh = wid >> 2;                // K half (keys 0-255 / 256-511)
            f32x8 acc = {};
            for (int kc = kh * 8; kc < kh * 8 + 8; ++kc) {
                Frag pa, vb;
                const unsigned short* pp = P + l15 * PSTRIDE + kc * 32 + half * 8;
                pa.q[0] = *(const U4*)pp;
                pa.q[1] = *(const U4*)(pp + 16);
                const unsigned short* vp = Vt + (ft * 16 + l15) * VSTRIDE + kc * 32 + half * 8;
                vb.q[0] = *(const U4*)vp;
                vb.q[1] = *(const U4*)(vp + 16);
                acc = __builtin_amdgcn_wmma_f32_16x16x32_bf16(false, pa.v, false, vb.v,
                                                              (short)0, acc, false, false);
            }
            float* pvs = (float*)(smem + S_OFF);    // S no longer needed
            if (kh == 1) {                          // wave-uniform
                #pragma unroll
                for (int r = 0; r < 8; ++r) pvs[(ft * 32 + lane) * 8 + r] = acc[r];
            }
            __syncthreads();
            if (kh == 0) {                          // wave-uniform
                #pragma unroll
                for (int r = 0; r < 8; ++r) {
                    int Mi = r + half * 8;
                    float o = (acc[r] + pvs[(ft * 32 + lane) * 8 + r]) * invs[Mi];
                    O[(size_t)(b * SEQ + l0 + Mi) * D_MODEL + h * DH + ft * 16 + l15] = o;
                }
            }
        }
        __syncthreads();
    }
}

// ===========================================================================
// In-place LayerNorm over last dim (768), one block per row.
// ===========================================================================
__global__ __launch_bounds__(256) void ln_kernel(
    float* __restrict__ x, const float* __restrict__ gamma, const float* __restrict__ beta)
{
    __shared__ float rs[256];
    __shared__ float mu_s, rstd_s;
    float* xr = x + (size_t)blockIdx.x * D_MODEL;
    const int tid = threadIdx.x;
    float v0 = xr[tid], v1 = xr[tid + 256], v2 = xr[tid + 512];
    rs[tid] = v0 + v1 + v2;
    __syncthreads();
    for (int o = 128; o > 0; o >>= 1) { if (tid < o) rs[tid] += rs[tid + o]; __syncthreads(); }
    if (tid == 0) mu_s = rs[0] * (1.f / D_MODEL);
    __syncthreads();
    float mu = mu_s;
    float d0 = v0 - mu, d1 = v1 - mu, d2 = v2 - mu;
    rs[tid] = d0 * d0 + d1 * d1 + d2 * d2;
    __syncthreads();
    for (int o = 128; o > 0; o >>= 1) { if (tid < o) rs[tid] += rs[tid + o]; __syncthreads(); }
    if (tid == 0) rstd_s = rsqrtf(rs[0] * (1.f / D_MODEL) + 1e-5f);
    __syncthreads();
    float r = rstd_s;
    xr[tid]       = d0 * r * gamma[tid]       + beta[tid];
    xr[tid + 256] = d1 * r * gamma[tid + 256] + beta[tid + 256];
    xr[tid + 512] = d2 * r * gamma[tid + 512] + beta[tid + 512];
}

// ===========================================================================
extern "C" void kernel_launch(void* const* d_in, const int* in_sizes, int n_in,
                              void* d_out, int out_size, void* d_ws, size_t ws_size,
                              hipStream_t stream)
{
    const float* q    = (const float*)d_in[0];
    const float* k    = (const float*)d_in[1];
    const float* v    = (const float*)d_in[2];
    const float* locs = (const float*)d_in[3];
    const float* txt  = (const float*)d_in[4];
    const float* Wq   = (const float*)d_in[5];
    const float* bq   = (const float*)d_in[6];
    const float* Wk   = (const float*)d_in[7];
    const float* bk   = (const float*)d_in[8];
    const float* Wv   = (const float*)d_in[9];
    const float* bv   = (const float*)d_in[10];
    const float* Wf   = (const float*)d_in[11];
    const float* bf_  = (const float*)d_in[12];
    const float* Wl   = (const float*)d_in[13];
    const float* bl   = (const float*)d_in[14];
    const float* gamma = (const float*)d_in[15];
    const float* beta  = (const float*)d_in[16];
    const unsigned char* mask = (const unsigned char*)d_in[17];

    const size_t BF_BYTES = (size_t)MROWS * D_MODEL * 2;   // 12,582,912
    char* ws = (char*)d_ws;
    unsigned short* Qb  = (unsigned short*)(ws);
    unsigned short* Kb  = (unsigned short*)(ws + BF_BYTES);
    unsigned short* Vb  = (unsigned short*)(ws + 2 * BF_BYTES);
    float*          swb = (float*)(ws + 3 * BF_BYTES);
    float*          O   = (float*)(ws + 3 * BF_BYTES + (size_t)MROWS * 72 * 4);

    (void)hipFuncSetAttribute((const void*)attn_kernel,
                              hipFuncAttributeMaxDynamicSharedMemorySize, SMEM_TOTAL);

    dim3 gg(D_MODEL / 64, MROWS / 64);       // 12 x 128

    proj_gemm_kernel<true,  false><<<gg, 256, 0, stream>>>(q, Wq, bq, nullptr, nullptr, Qb);
    proj_gemm_kernel<true,  false><<<gg, 256, 0, stream>>>(k, Wk, bk, nullptr, nullptr, Kb);
    proj_gemm_kernel<true,  false><<<gg, 256, 0, stream>>>(v, Wv, bv, nullptr, nullptr, Vb);

    sw_kernel<<<MROWS, 96, 0, stream>>>(q, txt, Wl, bl, swb);

    attn_kernel<<<dim3(SEQ / 16, BATCH), 256, SMEM_TOTAL, stream>>>(Qb, Kb, Vb, swb, locs, mask, O);

    proj_gemm_kernel<false, true><<<gg, 256, 0, stream>>>(O, Wf, bf_, q, (float*)d_out, nullptr);

    ln_kernel<<<MROWS, 256, 0, stream>>>((float*)d_out, gamma, beta);
}
